// CompositionBlock_31413390803005
// MI455X (gfx1250) — matrix operand order repaired and checked by reading, not compiled, not run on hardware
//
#include <hip/hip_runtime.h>
#include <hip/hip_bf16.h>

typedef __attribute__((ext_vector_type(16))) __bf16 v16bf;
typedef __attribute__((ext_vector_type(8)))  float  v8f;

union Frag16 {
    unsigned int u[8];
    uint4        q[2];
    v16bf        v;
    __bf16       h[16];
};

__device__ __forceinline__ unsigned short f2bf(float f) {
    union { __bf16 b; unsigned short s; } cv;
    cv.b = (__bf16)f;
    return cv.s;
}

// ---------------------------------------------------------------------------
// Kernel 1: out[b,i,t] = sum(w_red) * tanh(b_comp[t]) + b_red  (bias-only base)
// Each block cooperatively reduces sum(w_red) (S elems), then writes 256 outs.
// ---------------------------------------------------------------------------
__global__ __launch_bounds__(256)
void fill_base_kernel(const float* __restrict__ bcomp, const float* __restrict__ wred,
                      const float* __restrict__ bred, float* __restrict__ out,
                      int S, int T, long long total)
{
    __shared__ float red[256];
    float s = 0.0f;
    for (int i = threadIdx.x; i < S; i += 256) s += wred[i];
    red[threadIdx.x] = s;
    __syncthreads();
    for (int off = 128; off > 0; off >>= 1) {
        if (threadIdx.x < off) red[threadIdx.x] += red[threadIdx.x + off];
        __syncthreads();
    }
    float sumw = red[0];
    long long idx = (long long)blockIdx.x * 256 + threadIdx.x;
    if (idx < total) {
        int t = (int)(idx % T);
        out[idx] = sumw * tanhf(bcomp[t]) + bred[0];
    }
}

// ---------------------------------------------------------------------------
// Kernel 2: convert W_comp (f32 [T,T,T], T==128) into bf16 B-matrix fragment
// order so the main loop issues coalesced global_load_b128 directly into the
// v_wmma_f32_16x16x32_bf16 B operand layout.
// flat idx = ((((wt*128 + p)*4 + c)*2 + jh)*32 + lane)*8 + j'
//   t = wt*16 + (lane&15),  q = 32c + (lane>>4)*16 + jh*8 + j'
// ---------------------------------------------------------------------------
__global__ __launch_bounds__(256)
void conv_wfrag_kernel(const float* __restrict__ Wc, __bf16* __restrict__ wfrag, int total)
{
    int idx = blockIdx.x * 256 + threadIdx.x;
    if (idx >= total) return;
    int jp   = idx & 7;
    int lane = (idx >> 3) & 31;
    int jh   = (idx >> 8) & 1;
    int c    = (idx >> 9) & 3;
    int p    = (idx >> 11) & 127;
    int wt   = idx >> 18;
    int t    = wt * 16 + (lane & 15);
    int q    = (c << 5) + ((lane >> 4) << 4) + (jh << 3) + jp;
    wfrag[idx] = (__bf16)Wc[((size_t)t << 14) + (p << 7) + q];
}

// ---------------------------------------------------------------------------
// Kernel 3: main bilinear GEMM + epilogue scatter.
// Block = 256 threads (8 wave32); block handles 16 tokens x all 128 t.
// Wave wv handles t in [wv*16, wv*16+16). K = 16384 (p outer, q inner 4x32).
// ---------------------------------------------------------------------------
template<int USE_WFRAG>
__global__ __launch_bounds__(256)
void comp_main_kernel(const float* __restrict__ tok, const int* __restrict__ heads,
                      const float* __restrict__ Wc, const float* __restrict__ bcomp,
                      const float* __restrict__ wred, const __bf16* __restrict__ wfrag,
                      float* __restrict__ out, int S)
{
    __shared__ __bf16   hfragLds[4 * 32 * 16];  // A-layout tanh(x), 4 KB
    __shared__ unsigned xsLds[16 * 129];        // packed-bf16 x[s][p], padded

    const int tid = threadIdx.x;
    const int n0  = blockIdx.x << 4;  // first token of tile

    // xs: x[s,p] as (bf16<<16)|bf16 for v_pk_mul broadcast
    for (int i = tid; i < 16 * 128; i += 256) {
        int s = i >> 7, p = i & 127;
        unsigned short b = f2bf(tok[(size_t)(n0 + s) * 128 + p]);
        xsLds[s * 129 + p] = ((unsigned)b << 16) | b;
    }
    // hfrag: tanh(x[s,q]) pre-swizzled into this kernel's A-operand lane order (same k order as B)
    // lane half 0: K = {0..15}; half 1: K = {16..31} (per 32-chunk)
    for (int i = tid; i < 4 * 32 * 16; i += 256) {
        int c = i >> 9, rem = i & 511, L = rem >> 4, j = rem & 15;
        int hh = L >> 4, srow = L & 15;
        int q = (c << 5) + hh * 16 + j;
        float f = tanhf(tok[(size_t)(n0 + srow) * 128 + q]);
        hfragLds[(c * 32 + L) * 16 + j] = (__bf16)f;
    }
    __syncthreads();

    const int lane = tid & 31, wv = tid >> 5;
    const int half = lane >> 4, nloc = lane & 15;

    Frag16 hf[4];
    const uint4* hp = (const uint4*)hfragLds;
#pragma unroll
    for (int c = 0; c < 4; ++c) {
        hf[c].q[0] = hp[(c * 32 + lane) * 2 + 0];
        hf[c].q[1] = hp[(c * 32 + lane) * 2 + 1];
    }

    v8f acc;
#pragma unroll
    for (int r = 0; r < 8; ++r) acc[r] = 0.0f;

    const uint4* wb   = (const uint4*)wfrag + (size_t)wv * 32768;     // bf16 frag path
    const float* wrow = Wc + (size_t)(wv * 16 + nloc) * 16384;        // f32 fallback path

#pragma unroll 2
    for (int p = 0; p < 128; ++p) {
        unsigned xx = xsLds[nloc * 129 + p];  // x[s_row, p] packed twice
#pragma unroll
        for (int c = 0; c < 4; ++c) {
            // A fragment: u[s, q] = x[s,p] * h[s,q] via packed bf16 multiply
            Frag16 af;
#pragma unroll
            for (int k = 0; k < 8; ++k)
                asm("v_pk_mul_bf16 %0, %1, %2" : "=v"(af.u[k]) : "v"(xx), "v"(hf[c].u[k]));

            // B fragment: W[t, p, q] in B-matrix lane layout
            Frag16 bfv;
            if (USE_WFRAG) {
                size_t base = ((size_t)(p * 4 + c) * 2) * 32 + lane;
                bfv.q[0] = wb[base];
                bfv.q[1] = wb[base + 32];
            } else {
                const float4* w4 = (const float4*)(wrow + p * 128 + c * 32 + half * 16);
                float4 f0 = w4[0], f1 = w4[1], f2 = w4[2], f3 = w4[3];
                bfv.h[0]  = (__bf16)f0.x; bfv.h[1]  = (__bf16)f0.y;
                bfv.h[2]  = (__bf16)f0.z; bfv.h[3]  = (__bf16)f0.w;
                bfv.h[4]  = (__bf16)f1.x; bfv.h[5]  = (__bf16)f1.y;
                bfv.h[6]  = (__bf16)f1.z; bfv.h[7]  = (__bf16)f1.w;
                bfv.h[8]  = (__bf16)f2.x; bfv.h[9]  = (__bf16)f2.y;
                bfv.h[10] = (__bf16)f2.z; bfv.h[11] = (__bf16)f2.w;
                bfv.h[12] = (__bf16)f3.x; bfv.h[13] = (__bf16)f3.y;
                bfv.h[14] = (__bf16)f3.z; bfv.h[15] = (__bf16)f3.w;
            }
            acc = __builtin_amdgcn_wmma_f32_16x16x32_bf16(
                false, af.v, false, bfv.v, (short)0, acc, false, false);
        }
    }

    // Epilogue: act = tanh(val + b_comp); delta = w_red[s]*(act - tanh(b_comp));
    // scatter-add into out[b, dep_heads[b,s], t].
    const int t   = wv * 16 + nloc;
    float bc  = bcomp[t];
    float bvt = tanhf(bc);
    const int mbase = half * 8;  // C/D layout: lanes 16-31 hold M = r+8
#pragma unroll
    for (int r = 0; r < 8; ++r) {
        int n  = n0 + mbase + r;
        int b  = n / S;
        int sg = n - b * S;
        int hd = heads[(size_t)b * S + sg];
        float delta = wred[sg] * (tanhf(acc[r] + bc) - bvt);
        atomicAdd(out + ((size_t)b * S + hd) * 128 + t, delta);
    }
}

// ---------------------------------------------------------------------------
extern "C" void kernel_launch(void* const* d_in, const int* in_sizes, int n_in,
                              void* d_out, int out_size, void* d_ws, size_t ws_size,
                              hipStream_t stream)
{
    const float* tok   = (const float*)d_in[0];
    // d_in[1] (dep_embeddings) is dead per reference semantics
    const int*   heads = (const int*)d_in[2];
    const float* Wc    = (const float*)d_in[3];
    const float* bcomp = (const float*)d_in[4];
    const float* wred  = (const float*)d_in[5];
    const float* bred  = (const float*)d_in[6];
    float*       out   = (float*)d_out;

    const int T  = in_sizes[4];   // 128
    const int S  = in_sizes[5];   // 2048
    const int BS = in_sizes[2];   // B*S tokens
    if (T != 128) return;         // kernel tiling is specialized for T == 128

    const long long total = (long long)BS * T;
    fill_base_kernel<<<(int)((total + 255) / 256), 256, 0, stream>>>(
        bcomp, wred, bred, out, S, T, total);

    const int wfragElems = T * T * T;                 // 2,097,152 bf16
    const size_t wfragBytes = (size_t)wfragElems * 2; // 4 MB
    const int nTiles = BS / 16;

    if (ws_size >= wfragBytes) {
        __bf16* wfrag = (__bf16*)d_ws;
        conv_wfrag_kernel<<<(wfragElems + 255) / 256, 256, 0, stream>>>(Wc, wfrag, wfragElems);
        comp_main_kernel<1><<<nTiles, 256, 0, stream>>>(
            tok, heads, Wc, bcomp, wred, (const __bf16*)wfrag, out, S);
    } else {
        comp_main_kernel<0><<<nTiles, 256, 0, stream>>>(
            tok, heads, Wc, bcomp, wred, nullptr, out, S);
    }
}
